// AttnBlock_79577154060626
// MI455X (gfx1250) — matrix operand order repaired and checked by reading, not compiled, not run on hardware
//
#include <hip/hip_runtime.h>
#include <hip/hip_bf16.h>

// Problem constants (match reference)
#define BB 2
#define HH 96
#define WW 96
#define CC 96
#define NN (HH * WW)          // 9216 tokens per batch
#define GROUPS_ 32
#define HPG (HH / GROUPS_)    // 3 rows of H per group
#define GELEMS (HPG * WW * CC) // 27648 contiguous floats per group

typedef _Float16 h16;
typedef __attribute__((ext_vector_type(16))) _Float16 v16h;
typedef __attribute__((ext_vector_type(8)))  _Float16 v8h;
typedef __attribute__((ext_vector_type(8)))  float    v8f;
typedef __attribute__((ext_vector_type(4)))  float    v4f;

// q scale folded into Q at projection time: (1/sqrt(C)) * log2(e)
#define QSCALE 0.14724433f

#define WMMA16(A, Bm, Cm) \
  __builtin_amdgcn_wmma_f32_16x16x32_f16(false, (A), false, (Bm), (short)0, (Cm), false, false)

__device__ __forceinline__ v16h cat8(v8h lo, v8h hi) {
  v16h r;
#pragma unroll
  for (int i = 0; i < 8; ++i) { r[i] = lo[i]; r[i + 8] = hi[i]; }
  return r;
}

// ---------------------------------------------------------------------------
// Kernel 1: GroupNorm statistics. One block per (batch, group).
// Each group covers 3 consecutive H-rows -> 27648 contiguous floats.
// ---------------------------------------------------------------------------
__global__ __launch_bounds__(256) void gn_stats_kernel(const float* __restrict__ x,
                                                       float* __restrict__ stats) {
  const int gidx = blockIdx.x;            // 0..63
  const int b = gidx >> 5;
  const int g = gidx & 31;
  const float* base = x + ((size_t)b * HH + (size_t)g * HPG) * WW * CC;

  float s = 0.f, s2 = 0.f;
  for (int i = threadIdx.x; i < GELEMS; i += 256) {
    float v = base[i];
    s += v; s2 += v * v;
  }
#pragma unroll
  for (int off = 16; off >= 1; off >>= 1) {
    s  += __shfl_xor(s,  off, 32);
    s2 += __shfl_xor(s2, off, 32);
  }
  __shared__ float ls[8], ls2[8];
  if ((threadIdx.x & 31) == 0) { ls[threadIdx.x >> 5] = s; ls2[threadIdx.x >> 5] = s2; }
  __syncthreads();
  if (threadIdx.x == 0) {
    float S = 0.f, S2 = 0.f;
#pragma unroll
    for (int w = 0; w < 8; ++w) { S += ls[w]; S2 += ls2[w]; }
    float mean = S / (float)GELEMS;
    float var  = S2 / (float)GELEMS - mean * mean;
    stats[gidx * 2 + 0] = mean;
    stats[gidx * 2 + 1] = rsqrtf(var + 1e-5f);
  }
}

// ---------------------------------------------------------------------------
// Kernel 2: fused GroupNorm-normalize + Q/K/V projection.
// One block (96 threads) per token row. Writes:
//   Qh [B,N,C] f16  (pre-scaled by c^-0.5 * log2e)
//   Kh [B,N,C] f16
//   Vt [B,C,N] f16  (transposed for contiguous WMMA A-fragment loads)
// ---------------------------------------------------------------------------
__global__ __launch_bounds__(96) void qkv_kernel(
    const float* __restrict__ x,
    const float* __restrict__ gamma, const float* __restrict__ beta,
    const float* __restrict__ Wq, const float* __restrict__ bq,
    const float* __restrict__ Wk, const float* __restrict__ bk,
    const float* __restrict__ Wv, const float* __restrict__ bv,
    const float* __restrict__ stats,
    h16* __restrict__ Qh, h16* __restrict__ Kh, h16* __restrict__ Vt) {
  const int row = blockIdx.x;                   // 0 .. B*N-1
  const int b   = row / NN;
  const int rem = row - b * NN;                 // token index within batch
  const int hh  = rem / WW;                     // H coordinate (the GN "channel" axis)
  const int g   = hh / HPG;

  const float mean = stats[(b * 32 + g) * 2 + 0];
  const float rstd = stats[(b * 32 + g) * 2 + 1];
  const float gm = gamma[hh], bt = beta[hh];

  __shared__ float hn[CC];
  const int n = threadIdx.x;                    // output channel
  hn[n] = (x[(size_t)row * CC + n] - mean) * rstd * gm + bt;
  __syncthreads();

  float q = bq[n], k = bk[n], v = bv[n];
#pragma unroll 8
  for (int c = 0; c < CC; ++c) {
    float t = hn[c];
    q += t * Wq[c * CC + n];
    k += t * Wk[c * CC + n];
    v += t * Wv[c * CC + n];
  }
  Qh[(size_t)row * CC + n] = (h16)(q * QSCALE);
  Kh[(size_t)row * CC + n] = (h16)k;
  Vt[((size_t)b * CC + n) * NN + rem] = (h16)v;
}

// ---------------------------------------------------------------------------
// Online-softmax update for one 16-query tile from its two S^T fragments.
// Updates running max/sum, rescales the O^T accumulators, and returns the
// P^T B-fragment (32 keys x 16 queries) built via half-wave exchange.
// ---------------------------------------------------------------------------
__device__ __forceinline__ v16h softmax_update(v8f& st0, v8f& st1,
                                               float& mq, float& lq,
                                               v8f* of, int hgrp) {
  float tm = -3.0e38f;
#pragma unroll
  for (int r = 0; r < 8; ++r) tm = fmaxf(tm, fmaxf(st0[r], st1[r]));
  tm = fmaxf(tm, __shfl_xor(tm, 16, 32));
  float mnew  = fmaxf(mq, tm);
  float alpha = exp2f(mq - mnew);
  float rs = 0.f;
#pragma unroll
  for (int r = 0; r < 8; ++r) {
    st0[r] = exp2f(st0[r] - mnew);
    st1[r] = exp2f(st1[r] - mnew);
    rs += st0[r] + st1[r];
  }
  rs += __shfl_xor(rs, 16, 32);
  lq = lq * alpha + rs;
  mq = mnew;
#pragma unroll
  for (int m = 0; m < 6; ++m)
#pragma unroll
    for (int r = 0; r < 8; ++r) of[m][r] *= alpha;

  float snd[8], rmv[8];
#pragma unroll
  for (int r = 0; r < 8; ++r) snd[r] = hgrp ? st0[r] : st1[r];
#pragma unroll
  for (int r = 0; r < 8; ++r) rmv[r] = __shfl_xor(snd[r], 16, 32);
  v16h pB;
#pragma unroll
  for (int i = 0; i < 8; ++i) {
    float lo = hgrp ? rmv[i] : st0[i];   // keys 16h+0..7
    float hi = hgrp ? st1[i] : rmv[i];   // keys 16h+8..15
    pB[i]     = (h16)lo;
    pB[i + 8] = (h16)hi;
  }
  return pB;
}

// ---------------------------------------------------------------------------
// Kernel 3: flash attention with WMMA.
// grid = B * (N/128), block = 128 (4 waves). Wave owns 32 query rows
// (two 16-column tiles) so the K and V A-fragments are shared by both
// tiles: 24 WMMAs per 24 b128 loads per 32-key iteration.
// ---------------------------------------------------------------------------
__global__ __launch_bounds__(128) void flash_kernel(
    const h16* __restrict__ Qh, const h16* __restrict__ Kh,
    const h16* __restrict__ Vt, float* __restrict__ Ows) {
  const int lane = threadIdx.x & 31;
  const int wave = threadIdx.x >> 5;   // 0..3
  const int n    = lane & 15;          // fragment column (query for S^T, O^T)
  const int hgrp = lane >> 4;          // half-wave group

  const int blocksPerBatch = NN / 128;                 // 72
  const int b     = blockIdx.x / blocksPerBatch;
  const int qblk  = blockIdx.x % blocksPerBatch;
  const int qbase = qblk * 128 + wave * 32;

  const h16* Qb = Qh + (size_t)b * NN * CC;
  const h16* Kb = Kh + (size_t)b * NN * CC;
  const h16* Vb = Vt + (size_t)b * CC * NN;

  // Q^T B-fragments for both query tiles (contiguous c-chunk [32s+16h, +16))
  v16h qf0[3], qf1[3];
#pragma unroll
  for (int s = 0; s < 3; ++s) {
    qf0[s] = *(const v16h*)(Qb + (size_t)(qbase + n) * CC + 32 * s + 16 * hgrp);
    qf1[s] = *(const v16h*)(Qb + (size_t)(qbase + 16 + n) * CC + 32 * s + 16 * hgrp);
  }

  v8f of0[6], of1[6];
#pragma unroll
  for (int m = 0; m < 6; ++m)
#pragma unroll
    for (int r = 0; r < 8; ++r) { of0[m][r] = 0.f; of1[m][r] = 0.f; }

  float mq0 = -3.0e38f, lq0 = 0.f;
  float mq1 = -3.0e38f, lq1 = 0.f;

  for (int kt = 0; kt < NN; kt += 32) {
    // ---- S^T tiles: keys [kt,kt+16) and [kt+16,kt+32) x both query tiles
    v8f s00, s01, s10, s11;
#pragma unroll
    for (int r = 0; r < 8; ++r) { s00[r] = 0.f; s01[r] = 0.f; s10[r] = 0.f; s11[r] = 0.f; }
#pragma unroll
    for (int s = 0; s < 3; ++s) {
      const h16* kr0 = Kb + (size_t)(kt + n) * CC + 32 * s;
      v16h ka0 = *(const v16h*)(kr0 + 16 * hgrp);
      const h16* kr1 = Kb + (size_t)(kt + 16 + n) * CC + 32 * s;
      v16h ka1 = *(const v16h*)(kr1 + 16 * hgrp);
      s00 = WMMA16(ka0, qf0[s], s00);
      s01 = WMMA16(ka1, qf0[s], s01);
      s10 = WMMA16(ka0, qf1[s], s10);
      s11 = WMMA16(ka1, qf1[s], s11);
    }

    // ---- online softmax + P^T fragment for each query tile
    v16h pB0 = softmax_update(s00, s01, mq0, lq0, of0, hgrp);
    v16h pB1 = softmax_update(s10, s11, mq1, lq1, of1, hgrp);

    // ---- O^T += Vt_tile x P^T  (V fragment shared by both query tiles)
#pragma unroll
    for (int m = 0; m < 6; ++m) {
      const h16* vr = Vb + (size_t)(16 * m + n) * NN + kt;
      v16h va = *(const v16h*)(vr + 16 * hgrp);
      of0[m] = WMMA16(va, pB0, of0[m]);
      of1[m] = WMMA16(va, pB1, of1[m]);
    }
  }

  // ---- finalize and store O row-major [B,N,C] f32
  const float linv0 = 1.0f / lq0;
  const float linv1 = 1.0f / lq1;
  float* orow0 = Ows + ((size_t)b * NN + qbase + n) * CC;
  float* orow1 = Ows + ((size_t)b * NN + qbase + 16 + n) * CC;
#pragma unroll
  for (int m = 0; m < 6; ++m) {
    v4f a0, a1, c0, c1;
#pragma unroll
    for (int r = 0; r < 4; ++r) {
      a0[r] = of0[m][r] * linv0;  a1[r] = of0[m][r + 4] * linv0;
      c0[r] = of1[m][r] * linv1;  c1[r] = of1[m][r + 4] * linv1;
    }
    float* d0 = orow0 + 16 * m + 8 * hgrp;   // c = 16m + 8h + r contiguous in r
    float* d1 = orow1 + 16 * m + 8 * hgrp;
    *(v4f*)(d0)     = a0;
    *(v4f*)(d0 + 4) = a1;
    *(v4f*)(d1)     = c0;
    *(v4f*)(d1 + 4) = c1;
  }
}

// ---------------------------------------------------------------------------
// Kernel 4: out = x + O @ Wp + bp, via WMMA (f32->f16 fragment conversion
// on the fly). grid = B*N/128 blocks, 8 waves, wave owns 16 rows.
// ---------------------------------------------------------------------------
__global__ __launch_bounds__(256) void proj_kernel(
    const float* __restrict__ Ows, const float* __restrict__ x,
    const float* __restrict__ Wp, const float* __restrict__ bp,
    float* __restrict__ out) {
  const int lane = threadIdx.x & 31;
  const int wave = threadIdx.x >> 5;
  const int n    = lane & 15;
  const int hgrp = lane >> 4;
  const int rowbase = (blockIdx.x * 8 + wave) * 16;   // 0 .. B*N-16

  // A-fragments of O (rows = tokens, k = input channel)
  v16h af[3];
#pragma unroll
  for (int s = 0; s < 3; ++s) {
    const float* orow = Ows + (size_t)(rowbase + n) * CC + 32 * s;
    v4f lo0 = *(const v4f*)(orow + 8 * hgrp);
    v4f lo1 = *(const v4f*)(orow + 8 * hgrp + 4);
    v4f hi0 = *(const v4f*)(orow + 16 + 8 * hgrp);
    v4f hi1 = *(const v4f*)(orow + 16 + 8 * hgrp + 4);
    v16h a;
#pragma unroll
    for (int i = 0; i < 4; ++i) {
      a[i]      = (h16)lo0[i];
      a[i + 4]  = (h16)lo1[i];
      a[i + 8]  = (h16)hi0[i];
      a[i + 12] = (h16)hi1[i];
    }
    af[s] = a;
  }

#pragma unroll
  for (int nt = 0; nt < 6; ++nt) {
    v8f acc;
#pragma unroll
    for (int r = 0; r < 8; ++r) acc[r] = 0.f;
#pragma unroll
    for (int s = 0; s < 3; ++s) {
      v16h bf;
      const int col = 16 * nt + n;
#pragma unroll
      for (int i = 0; i < 16; ++i)
        bf[i] = (h16)Wp[(32 * s + 8 * hgrp + i + (i & 8)) * CC + col];
      acc = WMMA16(af[s], bf, acc);
    }
    const int col = 16 * nt + n;
    const float bias = bp[col];
#pragma unroll
    for (int r = 0; r < 8; ++r) {
      const size_t row = (size_t)rowbase + r + 8 * hgrp;
      out[row * CC + col] = x[row * CC + col] + bias + acc[r];
    }
  }
}

// ---------------------------------------------------------------------------
extern "C" void kernel_launch(void* const* d_in, const int* in_sizes, int n_in,
                              void* d_out, int out_size, void* d_ws, size_t ws_size,
                              hipStream_t stream) {
  const float* x     = (const float*)d_in[0];
  const float* gamma = (const float*)d_in[1];
  const float* beta  = (const float*)d_in[2];
  const float* Wq    = (const float*)d_in[3];
  const float* bq    = (const float*)d_in[4];
  const float* Wk    = (const float*)d_in[5];
  const float* bk    = (const float*)d_in[6];
  const float* Wv    = (const float*)d_in[7];
  const float* bv    = (const float*)d_in[8];
  const float* Wp    = (const float*)d_in[9];
  const float* bp    = (const float*)d_in[10];
  float* out = (float*)d_out;

  // Workspace layout (aligned to 256B):
  //   stats  : 64*2 f32
  //   Qh     : B*N*C f16
  //   Kh     : B*N*C f16
  //   Vt     : B*C*N f16
  //   Ows    : B*N*C f32
  char* w = (char*)d_ws;
  const size_t HALF_SZ = (size_t)BB * NN * CC * sizeof(h16);   // 3,538,944
  float* stats = (float*)(w);
  h16*   Qh    = (h16*)(w + 4096);
  h16*   Kh    = (h16*)(w + 4096 + HALF_SZ);
  h16*   Vt    = (h16*)(w + 4096 + 2 * HALF_SZ);
  float* Ows   = (float*)(w + 4096 + 3 * HALF_SZ);

  gn_stats_kernel<<<BB * GROUPS_, 256, 0, stream>>>(x, stats);

  qkv_kernel<<<BB * NN, CC, 0, stream>>>(x, gamma, beta,
                                         Wq, bq, Wk, bk, Wv, bv,
                                         stats, Qh, Kh, Vt);

  flash_kernel<<<BB * (NN / 128), 128, 0, stream>>>(Qh, Kh, Vt, Ows);

  proj_kernel<<<(BB * NN) / 128, 256, 0, stream>>>(Ows, x, Wp, bp, out);
}
